// SelfAttention_51900384805309
// MI455X (gfx1250) — matrix operand order repaired and checked by reading, not compiled, not run on hardware
//
#include <hip/hip_runtime.h>

// ---------------------------------------------------------------------------
// Types for CDNA5 WMMA (wave32): 16x16x32 f16 -> f32
// ---------------------------------------------------------------------------
typedef __attribute__((ext_vector_type(16))) _Float16 v16h;
typedef __attribute__((ext_vector_type(8)))  float    v8f;

union AFrag { v16h v; uint4 q[2]; };   // 32 bytes: per-lane A/B fragment
union BFrag { v16h v; uint4 q[2]; };

#define WMMA_F16(a, b, c) \
  __builtin_amdgcn_wmma_f32_16x16x32_f16(false, (a), false, (b), (short)0, (c), false, false)

// ---------------------------------------------------------------------------
// Problem constants
// ---------------------------------------------------------------------------
#define HIDDEN   1024
#define HEADS    16
#define HEAD_DIM 64
#define BATCH    4
#define SEQ      1024
#define M_TOT    (BATCH * SEQ)      // 4096 rows
#define EPSF     1e-5f

// ---------------------------------------------------------------------------
// Kernel 1: fp32 -> f16 conversion (weights)
// ---------------------------------------------------------------------------
__global__ void __launch_bounds__(256)
cvt_f32_f16(const float* __restrict__ src, _Float16* __restrict__ dst, int n) {
  int idx = (blockIdx.x * 256 + threadIdx.x) * 4;
  if (idx < n) {
    float4 v = *(const float4*)(src + idx);
    dst[idx + 0] = (_Float16)v.x;
    dst[idx + 1] = (_Float16)v.y;
    dst[idx + 2] = (_Float16)v.z;
    dst[idx + 3] = (_Float16)v.w;
  }
}

// ---------------------------------------------------------------------------
// Kernel 2: RMSNorm over HIDDEN, write f16. One block (256 thr) per row.
// ---------------------------------------------------------------------------
__global__ void __launch_bounds__(256)
rmsnorm_x(const float* __restrict__ x, const float* __restrict__ w,
          _Float16* __restrict__ h) {
  const int row = blockIdx.x;
  const int t   = threadIdx.x;
  const float* xr = x + (size_t)row * HIDDEN;
  float4 v = *(const float4*)(xr + t * 4);
  float ss = v.x * v.x + v.y * v.y + v.z * v.z + v.w * v.w;
  #pragma unroll
  for (int m = 1; m < 32; m <<= 1) ss += __shfl_xor(ss, m, 32);
  __shared__ float red[8];
  if ((t & 31) == 0) red[t >> 5] = ss;
  __syncthreads();
  float tot = 0.f;
  #pragma unroll
  for (int i = 0; i < 8; i++) tot += red[i];
  const float sc = rsqrtf(tot * (1.0f / HIDDEN) + EPSF);
  _Float16* hr = h + (size_t)row * HIDDEN;
  hr[t * 4 + 0] = (_Float16)(v.x * sc * w[t * 4 + 0]);
  hr[t * 4 + 1] = (_Float16)(v.y * sc * w[t * 4 + 1]);
  hr[t * 4 + 2] = (_Float16)(v.z * sc * w[t * 4 + 2]);
  hr[t * 4 + 3] = (_Float16)(v.w * sc * w[t * 4 + 3]);
}

// ---------------------------------------------------------------------------
// Kernel 3/6: WMMA GEMM  C[M,N] = A[M,K] * Bw[N,K]^T + bias
//   block = 128 threads = 4 waves in a 2x2 grid; wave tile = 64x64
//   A,Bw are K-contiguous f16 -> fragment loads are contiguous b128s.
// ---------------------------------------------------------------------------
__global__ void __launch_bounds__(128)
gemm_f16(const _Float16* __restrict__ A, const _Float16* __restrict__ Bw,
         const float* __restrict__ bias, float* __restrict__ C,
         int M, int N, int K) {
  const int lane  = threadIdx.x & 31;
  const int wave  = threadIdx.x >> 5;
  const int lm    = lane & 15;
  const int hi    = lane >> 4;
  const int mBase = blockIdx.y * 128 + (wave >> 1) * 64;
  const int nBase = blockIdx.x * 128 + (wave & 1) * 64;

  v8f acc[4][4];
  #pragma unroll
  for (int i = 0; i < 4; i++)
    #pragma unroll
    for (int j = 0; j < 4; j++)
      #pragma unroll
      for (int r = 0; r < 8; r++) acc[i][j][r] = 0.f;

  for (int kk = 0; kk < K; kk += 32) {
    AFrag af[4];
    BFrag bf[4];
    #pragma unroll
    for (int i = 0; i < 4; i++) {
      const _Float16* ap = A + (size_t)(mBase + 16 * i + lm) * K + kk;
      af[i].q[0] = *(const uint4*)(ap + 8 * hi);        // K = 8*hi .. +7
      af[i].q[1] = *(const uint4*)(ap + 16 + 8 * hi);   // K = 16+8*hi .. +7
    }
    #pragma unroll
    for (int j = 0; j < 4; j++) {
      const _Float16* bp = Bw + (size_t)(nBase + 16 * j + lm) * K + kk;
      bf[j].q[0] = *(const uint4*)(bp + 8 * hi);        // K = 8*hi .. +7
      bf[j].q[1] = *(const uint4*)(bp + 16 + 8 * hi);   // K = 16+8*hi .. +7
    }
    #pragma unroll
    for (int i = 0; i < 4; i++)
      #pragma unroll
      for (int j = 0; j < 4; j++)
        acc[i][j] = WMMA_F16(af[i].v, bf[j].v, acc[i][j]);
  }

  #pragma unroll
  for (int j = 0; j < 4; j++) {
    const int n  = nBase + 16 * j + lm;
    const float bv = bias[n];
    #pragma unroll
    for (int i = 0; i < 4; i++)
      #pragma unroll
      for (int r = 0; r < 8; r++) {
        const int m = mBase + 16 * i + 8 * hi + r;
        C[(size_t)m * N + n] = acc[i][j][r] + bv;
      }
  }
}

// ---------------------------------------------------------------------------
// Kernel 4: per-head qk RMSNorm + reshape to [bh, s, d] f16; V transposed to
//           [bh, d, s] f16 so the attention P*V B-fragments are contiguous.
// Block = 128 thr (4 waves) handles one (b,s) row of qkv[4096,3072].
// ---------------------------------------------------------------------------
__global__ void __launch_bounds__(128)
qk_norm_v(const float* __restrict__ qkv, const float* __restrict__ qw,
          const float* __restrict__ kw, _Float16* __restrict__ q16,
          _Float16* __restrict__ k16, _Float16* __restrict__ vt16) {
  const int bs   = blockIdx.x;            // b*SEQ + s
  const int b    = bs >> 10;
  const int s    = bs & 1023;
  const int wave = threadIdx.x >> 5;
  const int lane = threadIdx.x & 31;
  const float* row = qkv + (size_t)bs * (3 * HIDDEN);

  #pragma unroll
  for (int hh = 0; hh < 4; hh++) {
    const int h  = wave * 4 + hh;
    const int bh = b * HEADS + h;
    // ---- q ----
    {
      const float* p = row + h * HEAD_DIM;
      float a = p[lane], c = p[lane + 32];
      float ss = a * a + c * c;
      #pragma unroll
      for (int m = 1; m < 32; m <<= 1) ss += __shfl_xor(ss, m, 32);
      const float sc = rsqrtf(ss * (1.0f / HEAD_DIM) + EPSF);
      const size_t base = ((size_t)bh * SEQ + s) * HEAD_DIM;
      q16[base + lane]      = (_Float16)(a * sc * qw[lane]);
      q16[base + lane + 32] = (_Float16)(c * sc * qw[lane + 32]);
    }
    // ---- k ----
    {
      const float* p = row + HIDDEN + h * HEAD_DIM;
      float a = p[lane], c = p[lane + 32];
      float ss = a * a + c * c;
      #pragma unroll
      for (int m = 1; m < 32; m <<= 1) ss += __shfl_xor(ss, m, 32);
      const float sc = rsqrtf(ss * (1.0f / HEAD_DIM) + EPSF);
      const size_t base = ((size_t)bh * SEQ + s) * HEAD_DIM;
      k16[base + lane]      = (_Float16)(a * sc * kw[lane]);
      k16[base + lane + 32] = (_Float16)(c * sc * kw[lane + 32]);
    }
    // ---- v (transpose to [bh, d, s]) ----
    {
      const float* p = row + 2 * HIDDEN + h * HEAD_DIM;
      const size_t vb = (size_t)bh * HEAD_DIM * SEQ + s;
      vt16[vb + (size_t)lane * SEQ]        = (_Float16)p[lane];
      vt16[vb + (size_t)(lane + 32) * SEQ] = (_Float16)p[lane + 32];
    }
  }
}

// ---------------------------------------------------------------------------
// Kernel 5: attention for one (bh, 64-query tile) per block; 4 waves, each
// wave owns 16 query rows. Streams keys in chunks of 32 with online softmax.
// ---------------------------------------------------------------------------
__global__ void __launch_bounds__(128)
attn(const _Float16* __restrict__ q16, const _Float16* __restrict__ k16,
     const _Float16* __restrict__ vt16, _Float16* __restrict__ o16) {
  const int bh   = blockIdx.y;            // 0..63
  const int b    = bh >> 4;
  const int h    = bh & 15;
  const int wave = threadIdx.x >> 5;
  const int lane = threadIdx.x & 31;
  const int lm   = lane & 15;
  const int hi   = lane >> 4;
  const int qBase = blockIdx.x * 64 + wave * 16;

  const _Float16* qp = q16  + (size_t)bh * SEQ * HEAD_DIM;
  const _Float16* kp = k16  + (size_t)bh * SEQ * HEAD_DIM;
  const _Float16* vp = vt16 + (size_t)bh * HEAD_DIM * SEQ;

  // Q A-fragments for the two 32-wide K chunks of head_dim=64 (kept in regs)
  AFrag qa[2];
  #pragma unroll
  for (int c = 0; c < 2; c++) {
    const _Float16* ap = qp + (size_t)(qBase + lm) * HEAD_DIM + 32 * c;
    qa[c].q[0] = *(const uint4*)(ap + 8 * hi);
    qa[c].q[1] = *(const uint4*)(ap + 16 + 8 * hi);
  }

  float mrow[8], lrow[8];
  #pragma unroll
  for (int r = 0; r < 8; r++) { mrow[r] = -1e30f; lrow[r] = 0.f; }
  v8f o[4];
  #pragma unroll
  for (int j = 0; j < 4; j++)
    #pragma unroll
    for (int r = 0; r < 8; r++) o[j][r] = 0.f;

  __shared__ __align__(16) _Float16 pTile[4][16][32];   // per-wave P tile
  const float scale = 0.125f;                            // 1/sqrt(64)

  for (int j0 = 0; j0 < SEQ; j0 += 32) {
    // ---- S = Q K^T for 32 keys (two 16x16 tiles) ----
    v8f sf[2];
    #pragma unroll
    for (int c = 0; c < 2; c++) {
      #pragma unroll
      for (int r = 0; r < 8; r++) sf[c][r] = 0.f;
      #pragma unroll
      for (int kk = 0; kk < 2; kk++) {
        BFrag bf;
        const _Float16* bp =
            kp + (size_t)(j0 + 16 * c + lm) * HEAD_DIM + 32 * kk;
        bf.q[0] = *(const uint4*)(bp + 8 * hi);
        bf.q[1] = *(const uint4*)(bp + 16 + 8 * hi);
        sf[c] = WMMA_F16(qa[kk].v, bf.v, sf[c]);
      }
    }
    // ---- online softmax (row = qBase + 8*hi + r; 16 lanes per row) ----
    float corr[8];
    #pragma unroll
    for (int r = 0; r < 8; r++) {
      float s0 = sf[0][r] * scale, s1 = sf[1][r] * scale;
      float mx = fmaxf(s0, s1);
      #pragma unroll
      for (int msk = 1; msk < 16; msk <<= 1)
        mx = fmaxf(mx, __shfl_xor(mx, msk, 32));
      const float mnew = fmaxf(mrow[r], mx);
      const float p0 = __expf(s0 - mnew), p1 = __expf(s1 - mnew);
      float rs = p0 + p1;
      #pragma unroll
      for (int msk = 1; msk < 16; msk <<= 1) rs += __shfl_xor(rs, msk, 32);
      corr[r] = __expf(mrow[r] - mnew);
      lrow[r] = lrow[r] * corr[r] + rs;
      mrow[r] = mnew;
      sf[0][r] = p0;
      sf[1][r] = p1;
    }
    #pragma unroll
    for (int j = 0; j < 4; j++)
      #pragma unroll
      for (int r = 0; r < 8; r++) o[j][r] *= corr[r];

    // ---- C-layout P -> LDS -> A-layout fragment ----
    __syncthreads();                       // WAR vs previous iteration's reads
    #pragma unroll
    for (int r = 0; r < 8; r++) {
      pTile[wave][8 * hi + r][lm]      = (_Float16)sf[0][r];
      pTile[wave][8 * hi + r][16 + lm] = (_Float16)sf[1][r];
    }
    __syncthreads();                       // RAW before fragment read
    AFrag pa;
    const _Float16* pp = &pTile[wave][lm][0];
    pa.q[0] = *(const uint4*)(pp + 8 * hi);
    pa.q[1] = *(const uint4*)(pp + 16 + 8 * hi);

    // ---- O += P * V  (V transposed: contiguous B-fragments) ----
    #pragma unroll
    for (int j = 0; j < 4; j++) {
      BFrag vb;
      const _Float16* bp = vp + (size_t)(16 * j + lm) * SEQ + j0;
      vb.q[0] = *(const uint4*)(bp + 8 * hi);
      vb.q[1] = *(const uint4*)(bp + 16 + 8 * hi);
      o[j] = WMMA_F16(pa.v, vb.v, o[j]);
    }
  }

  // ---- finalize: O /= l ; write o16 as [b, s, h*64+d] ----
  #pragma unroll
  for (int r = 0; r < 8; r++) lrow[r] = 1.0f / lrow[r];
  #pragma unroll
  for (int j = 0; j < 4; j++)
    #pragma unroll
    for (int r = 0; r < 8; r++) {
      const int s = qBase + 8 * hi + r;
      o16[((size_t)(b * SEQ + s)) * HIDDEN + h * HEAD_DIM + 16 * j + lm] =
          (_Float16)(o[j][r] * lrow[r]);
    }
}

// ---------------------------------------------------------------------------
// Host-side launcher
// ---------------------------------------------------------------------------
extern "C" void kernel_launch(void* const* d_in, const int* in_sizes, int n_in,
                              void* d_out, int out_size, void* d_ws, size_t ws_size,
                              hipStream_t stream) {
  const float* x      = (const float*)d_in[0];
  const float* w_in   = (const float*)d_in[1];
  const float* b_in   = (const float*)d_in[2];
  const float* w_norm = (const float*)d_in[3];
  const float* w_out  = (const float*)d_in[4];
  const float* b_out  = (const float*)d_in[5];
  const float* q_nw   = (const float*)d_in[6];
  const float* k_nw   = (const float*)d_in[7];
  float* out = (float*)d_out;

  char* ws = (char*)d_ws;
  const size_t MB = 1024 * 1024;
  _Float16* h16    = (_Float16*)(ws + 0 * MB);    // [4096,1024]  8 MB
  _Float16* win16  = (_Float16*)(ws + 8 * MB);    // [3072,1024]  6 MB
  _Float16* wout16 = (_Float16*)(ws + 14 * MB);   // [1024,1024]  2 MB
  float*    qkv32  = (float*)   (ws + 16 * MB);   // [4096,3072] 48 MB
  _Float16* q16    = (_Float16*)(ws + 64 * MB);   // [64,1024,64] 8 MB
  _Float16* k16    = (_Float16*)(ws + 72 * MB);   // [64,1024,64] 8 MB
  _Float16* vt16   = (_Float16*)(ws + 80 * MB);   // [64,64,1024] 8 MB
  _Float16* o16    = (_Float16*)(ws + 88 * MB);   // [4096,1024]  8 MB

  // 1. weights -> f16
  cvt_f32_f16<<<(3 * HIDDEN * HIDDEN / 4) / 256, 256, 0, stream>>>(
      w_in, win16, 3 * HIDDEN * HIDDEN);
  cvt_f32_f16<<<(HIDDEN * HIDDEN / 4) / 256, 256, 0, stream>>>(
      w_out, wout16, HIDDEN * HIDDEN);
  // 2. pre-norm -> f16
  rmsnorm_x<<<M_TOT, 256, 0, stream>>>(x, w_norm, h16);
  // 3. QKV GEMM (WMMA): [4096,3072]
  gemm_f16<<<dim3(3 * HIDDEN / 128, M_TOT / 128), 128, 0, stream>>>(
      h16, win16, b_in, qkv32, M_TOT, 3 * HIDDEN, HIDDEN);
  // 4. qk-norm + reshape + V transpose
  qk_norm_v<<<M_TOT, 128, 0, stream>>>(qkv32, q_nw, k_nw, q16, k16, vt16);
  // 5. attention (WMMA flash-style)
  attn<<<dim3(SEQ / 64, BATCH * HEADS), 128, 0, stream>>>(q16, k16, vt16, o16);
  // 6. output projection (WMMA) -> fp32 d_out
  gemm_f16<<<dim3(HIDDEN / 128, M_TOT / 128), 128, 0, stream>>>(
      o16, wout16, b_out, out, M_TOT, HIDDEN, HIDDEN);
}
